// FNO_RC_51376398795190
// MI455X (gfx1250) — matrix operand order repaired and checked by reading, not compiled, not run on hardware
//
#include <hip/hip_runtime.h>
#include <hip/hip_bf16.h>

// ---------------------------------------------------------------------------
// FNO-RC forward for MI455X (gfx1250).
//
// All matrix contractions (truncated DFTs, per-mode channel mix, pointwise
// conv, fc1) run through v_wmma_f32_16x16x32_bf16 (bf16 operands / f32 acc).
// Wave tile = 16 x (NT*16), NT a template parameter (1/2/4) picked from N so
// the K-loop has NO conditionals: fixed-register accumulators, full EXEC at
// every WMMA.  Tiles are staged into LDS in *fragment order* so fragment
// loads are two ds_load_b128 per operand; staging is two-phase (8 loads in
// flight, then convert+ds_store_b32 of packed bf16x2); bounds guards are
// branchless (clamped address + select).
//
// Input pointer order assumption (dict insertion order of setup_inputs /
// _make_params): d_in[0]=x, 1..6 = fc0_w, fc0_b, fc1_w, fc1_b, fc2_w, fc2_b,
// then per layer i: 7+8i..14+8i = w1, w2, g_w1, g_b1, g_w2, g_b2, w{i}_w, w{i}_b.
//
// Workspace requirement: ~310 MB (layout below).
// ---------------------------------------------------------------------------

typedef __attribute__((ext_vector_type(8)))  float  v8f;
typedef __attribute__((ext_vector_type(16))) __bf16 v16bf;
typedef __attribute__((ext_vector_type(2)))  __bf16 v2bf;

#define NB   8      // batch
#define NC   32     // channels (WIDTH)
#define HH   256    // padded H
#define WW   256    // padded W
#define HW   65536
#define M2   16     // ky modes
#define KXN  32     // kx modes kept (0..15, -16..-1)
#define SX   247
#define SY   247

__device__ __forceinline__ float gelu_f(float x) {
    return 0.5f * x * (1.0f + erff(x * 0.70710678118654752f));
}

// ---------------------------------------------------------------------------
// Core batched GEMM:  C = alpha*A*B (+C if acc) (+bias[m]) (gelu if act==1)
// A: M x K (row major, lda), B: K x N (row major, ldb), C: M x N (ldc).
// Wave tile: 16(M) x NT*16(N); K stepped by 32.  K must be a multiple of 32.
//
// LDS fragment-order layouts (per CDNA5 WMMA VGPR mapping):
//  A (16x32):  frag lane L holds m=L&15, kgroup=L>>4; element e -> k =
//              ((e>>3)<<4) + (kgroup<<3) + (e&7).  Inverse used for staging:
//              dst = (m + ((k>>3)&1)*16)*16 + ((k>>4)<<3) + (k&7).
//  B (32x16):  frag lane L holds n=L&15, kgroup=L>>4; element e -> k =
//              ((e>>3)<<4) + (kgroup<<3) + (e&7).  Inverse: dst = t*512 +
//              ((n&15) + ((k>>3)&1)*16)*16 + ((k>>4)<<3) + (k&7)   (t = n>>4 subtile).
// Even-k pairs are always contiguous in both layouts -> bf16x2 stores.
// ---------------------------------------------------------------------------
template <typename TA, typename TB, typename TC, int NT>
__global__ void gemm_wmma_kernel(const TA* __restrict__ Ag,
                                 const TB* __restrict__ Bg,
                                 TC* __restrict__ Cg,
                                 const float* __restrict__ bias,
                                 int M, int N, int K,
                                 int lda, int ldb, int ldc,
                                 long long sA, long long sB, long long sC,
                                 int ntm, int ntn, int nbatch,
                                 float alpha, int acc, int act) {
    constexpr int BN   = NT * 16;                 // tile width
    constexpr int BSH  = (NT == 1) ? 4 : ((NT == 2) ? 5 : 6);   // log2(BN)
    constexpr int PB   = 16 * BN / 32;            // B pairs per lane (8/16/32)
    constexpr int WSL  = 512 + NT * 512;          // LDS bf16 per wave
    __shared__ __bf16 lds[4 * WSL];
    const int lane = threadIdx.x & 31;
    const int wave = threadIdx.x >> 5;
    const long long tile = (long long)blockIdx.x * 4 + wave;
    const long long total = (long long)nbatch * ntm * ntn;
    if (tile >= total) return;                    // whole-wave uniform exit

    const int batch = (int)(tile / ((long long)ntm * ntn));
    const int rem   = (int)(tile % ((long long)ntm * ntn));
    const int m0 = (rem / ntn) * 16;
    const int n0 = (rem % ntn) * BN;

    const TA* A  = Ag + (long long)batch * sA;
    const TB* Bp = Bg + (long long)batch * sB;
    TC* C = Cg + (long long)batch * sC;

    __bf16* la = &lds[wave * WSL];
    __bf16* lb = la + 512;
    const int mlast = M - 1, nlast = N - 1;

    v8f accv[NT] = {};
    for (int k0 = 0; k0 < K; k0 += 32) {
        // ---- stage A (16x32): phase 1 issues 8 loads, phase 2 converts ----
        {
            float a0[8], a1[8];
#pragma unroll
            for (int j = 0; j < 8; ++j) {
                int i = lane + 32 * j;
                int m = i >> 4, kp = (i & 15) << 1;
                int mm = m0 + m;
                int msrc = (mm < M) ? mm : mlast;        // clamped, always valid
                const TA* ap = A + (long long)msrc * lda + (k0 + kp);
                a0[j] = (float)ap[0];
                a1[j] = (float)ap[1];
            }
#pragma unroll
            for (int j = 0; j < 8; ++j) {
                int i = lane + 32 * j;
                int m = i >> 4, kp = (i & 15) << 1;
                int mm = m0 + m;
                float v0 = (mm < M) ? a0[j] : 0.0f;      // select, no branch
                float v1 = (mm < M) ? a1[j] : 0.0f;
                v2bf p; p.x = (__bf16)v0; p.y = (__bf16)v1;
                int dst = ((m + (((kp >> 3) & 1) << 4)) << 4) + ((kp >> 4) << 3) + (kp & 7);
                *(v2bf*)(&la[dst]) = p;
            }
        }
        // ---- stage B (32 x BN): chunks of 8 pairs, loads pipelined ----
#pragma unroll
        for (int ch = 0; ch < PB; ch += 8) {
            float b0[8], b1[8];
#pragma unroll
            for (int j = 0; j < 8; ++j) {
                int i = lane + 32 * (ch + j);
                int n = i & (BN - 1), kp = (i >> BSH) << 1;
                int nn = n0 + n;
                int nsrc = (nn < N) ? nn : nlast;
                const TB* bp = Bp + (long long)(k0 + kp) * ldb + nsrc;
                b0[j] = (float)bp[0];
                b1[j] = (float)bp[ldb];
            }
#pragma unroll
            for (int j = 0; j < 8; ++j) {
                int i = lane + 32 * (ch + j);
                int n = i & (BN - 1), kp = (i >> BSH) << 1;
                int nn = n0 + n;
                float v0 = (nn < N) ? b0[j] : 0.0f;
                float v1 = (nn < N) ? b1[j] : 0.0f;
                v2bf p; p.x = (__bf16)v0; p.y = (__bf16)v1;
                int dst = ((n >> 4) << 9) + (((n & 15) + (((kp >> 3) & 1) << 4)) << 4) + ((kp >> 4) << 3) + (kp & 7);
                *(v2bf*)(&lb[dst]) = p;
            }
        }
        // prefetch next K panel of A (gfx1250 global_prefetch_b8)
        if (k0 + 32 < K) {
            int mp = m0 + (lane & 15);
            mp = (mp < M) ? mp : mlast;
            __builtin_prefetch((const void*)&A[(long long)mp * lda + k0 + 32], 0, 1);
        }
        // ---- fragments: contiguous 32B per lane -> 2x ds_load_b128 each ----
        v16bf af = *(const v16bf*)(la + (lane << 4));
#pragma unroll
        for (int t = 0; t < NT; ++t) {
            v16bf bf_ = *(const v16bf*)(lb + (t << 9) + (lane << 4));
            accv[t] = __builtin_amdgcn_wmma_f32_16x16x32_bf16(
                false, af, false, bf_, (short)0, accv[t], false, false);
        }
    }
    // ---- store: C/D layout, VGPR r -> m = r + 8*(lane>>4), n = lane&15 ----
#pragma unroll
    for (int t = 0; t < NT; ++t) {
#pragma unroll
        for (int r = 0; r < 8; ++r) {
            int m = m0 + r + ((lane >> 4) << 3);
            int n = n0 + t * 16 + (lane & 15);
            if (m < M && n < N) {
                float v = alpha * accv[t][r];
                if (acc)  v += (float)C[(long long)m * ldc + n];
                if (bias) v += bias[m];
                if (act == 1) v = gelu_f(v);
                C[(long long)m * ldc + n] = (TC)v;
            }
        }
    }
}

template <typename TA, typename TB, typename TC>
static void gemm(hipStream_t s, const TA* A, const TB* B, TC* C, const float* bias,
                 int M, int N, int K, int lda, int ldb, int ldc,
                 long long sA, long long sB, long long sC, int nbatch,
                 float alpha, int acc, int act) {
    int ntm = (M + 15) / 16;
    if (N <= 16) {
        int ntn = (N + 15) / 16;
        long long total = (long long)nbatch * ntm * ntn;
        gemm_wmma_kernel<TA, TB, TC, 1><<<(int)((total + 3) / 4), 128, 0, s>>>(
            A, B, C, bias, M, N, K, lda, ldb, ldc, sA, sB, sC, ntm, ntn, nbatch,
            alpha, acc, act);
    } else if (N <= 32) {
        int ntn = (N + 31) / 32;
        long long total = (long long)nbatch * ntm * ntn;
        gemm_wmma_kernel<TA, TB, TC, 2><<<(int)((total + 3) / 4), 128, 0, s>>>(
            A, B, C, bias, M, N, K, lda, ldb, ldc, sA, sB, sC, ntm, ntn, nbatch,
            alpha, acc, act);
    } else {
        int ntn = (N + 63) / 64;
        long long total = (long long)nbatch * ntm * ntn;
        gemm_wmma_kernel<TA, TB, TC, 4><<<(int)((total + 3) / 4), 128, 0, s>>>(
            A, B, C, bias, M, N, K, lda, ldb, ldc, sA, sB, sC, ntm, ntn, nbatch,
            alpha, acc, act);
    }
}

// ---------------------------------------------------------------------------
// Setup kernel: DFT / inverse-DFT / CFT constant matrices (single block).
// ---------------------------------------------------------------------------
__global__ void setup_consts_kernel(float* __restrict__ khr, float* __restrict__ khi,
                                    __bf16* __restrict__ fwr, __bf16* __restrict__ fwi,
                                    __bf16* __restrict__ fhr, __bf16* __restrict__ fhi,
                                    __bf16* __restrict__ ifhr, __bf16* __restrict__ ifhi,
                                    __bf16* __restrict__ iwb, __bf16* __restrict__ kw) {
    __shared__ float Bm[8][64];
    __shared__ float P[8][64];
    __shared__ float G[2][4][4][8][2];
    __shared__ float aug[8][72];
    const int tid = threadIdx.x;   // 256 threads

    // Chebyshev basis T_m(u) on 64 segment points
    for (int i = tid; i < 512; i += 256) {
        int m = i >> 6, s = i & 63;
        float u = -1.0f + 2.0f * s / 63.0f;
        float t0 = 1.0f, t1 = u, v = (m == 0) ? 1.0f : u;
        for (int q = 2; q <= m; ++q) { v = 2.0f * u * t1 - t0; t0 = t1; t1 = v; }
        Bm[m][s] = v;
    }
    __syncthreads();
    if (tid == 0) {
        // P = (Bm Bm^T)^{-1} Bm via Gauss-Jordan on [BBT | Bm]
        for (int i = 0; i < 8; ++i)
            for (int j = 0; j < 72; ++j) {
                if (j < 8) {
                    float sum = 0.0f;
                    for (int k = 0; k < 64; ++k) sum += Bm[i][k] * Bm[j][k];
                    aug[i][j] = sum;
                } else aug[i][j] = Bm[i][j - 8];
            }
        for (int p = 0; p < 8; ++p) {
            float inv = 1.0f / aug[p][p];
            for (int j = 0; j < 72; ++j) aug[p][j] *= inv;
            for (int r = 0; r < 8; ++r) if (r != p) {
                float f = aug[r][p];
                for (int j = 0; j < 72; ++j) aug[r][j] -= f * aug[p][j];
            }
        }
        for (int m = 0; m < 8; ++m)
            for (int s = 0; s < 64; ++s) P[m][s] = aug[m][8 + s];
    }
    __syncthreads();
    // G[kern][f][l][m] = sum_s Bm[m][s] e^{-2pi i f t(l,s)}
    {
        const int kern = tid >> 7, rem = tid & 127;
        const int f = rem >> 5, l = (rem >> 3) & 3, m = rem & 7;
        const int fhtab[4] = {0, 1, -2, -1};
        float fr = kern ? (float)fhtab[f] : (float)f;
        float re = 0.0f, im = 0.0f;
        for (int s = 0; s < 64; ++s) {
            float t = (l * 64 + s) * (1.0f / 255.0f);
            float ang = -6.283185307f * fr * t;
            re += Bm[m][s] * cosf(ang);
            im += Bm[m][s] * sinf(ang);
        }
        G[kern][f][l][m][0] = re; G[kern][f][l][m][1] = im;
    }
    __syncthreads();
    const float dt = 1.0f / 255.0f;
    for (int idx = tid; idx < 2048; idx += 256) {
        int kern = idx >> 10, f = (idx >> 8) & 3, t = idx & 255;
        int l = t >> 6, sp = t & 63;
        float re = 0.0f, im = 0.0f;
        for (int m = 0; m < 8; ++m) {
            re += P[m][sp] * G[kern][f][l][m][0];
            im += P[m][sp] * G[kern][f][l][m][1];
        }
        re *= dt; im *= dt;
        if (kern == 0) { kw[t * 8 + f] = (__bf16)re; kw[t * 8 + 4 + f] = (__bf16)im; }
        else           { khr[f * 256 + t] = re;      khi[f * 256 + t] = im; }
    }
    // forward DFT along w: Fw[t][ky] = e^{-2pi i ky t / 256}
    for (int idx = tid; idx < 4096; idx += 256) {
        int t = idx >> 4, ky = idx & 15;
        float ang = -6.283185307f * ky * t / 256.0f;
        fwr[idx] = (__bf16)cosf(ang); fwi[idx] = (__bf16)sinf(ang);
    }
    // forward DFT along h (32 kept kx): Fh[j][h]
    for (int idx = tid; idx < 8192; idx += 256) {
        int j = idx >> 8, h = idx & 255;
        float fr = (j < 16) ? (float)j : (float)(j - 32);
        float ang = -6.283185307f * fr * h / 256.0f;
        fhr[idx] = (__bf16)cosf(ang); fhi[idx] = (__bf16)sinf(ang);
    }
    // inverse along h: IFh[h][j] = e^{+2pi i f_j h / 256}
    for (int idx = tid; idx < 8192; idx += 256) {
        int h = idx >> 5, j = idx & 31;
        float fr = (j < 16) ? (float)j : (float)(j - 32);
        float ang = 6.283185307f * fr * h / 256.0f;
        ifhr[idx] = (__bf16)cosf(ang); ifhi[idx] = (__bf16)sinf(ang);
    }
    // inverse along w with rfft symmetry + 1/(H*W) folded:
    // rows 0..15 (real part): a(ky)/HW * cos; rows 16..31 (imag): -a/HW * sin
    for (int idx = tid; idx < 8192; idx += 256) {
        int row = idx >> 8, w = idx & 255;
        int ky = row & 15;
        float a = (ky == 0) ? 1.0f : 2.0f;
        float sc = a / 65536.0f;
        float ang = 6.283185307f * ky * w / 256.0f;
        iwb[idx] = (__bf16)((row < 16) ? sc * cosf(ang) : -sc * sinf(ang));
    }
}

// pack spectral weights -> bf16 [kx(32)][ky(16)][c(32)][o(32)] re/im planes
__global__ void pack_w_kernel(const float* __restrict__ w1, const float* __restrict__ w2,
                              __bf16* __restrict__ wr, __bf16* __restrict__ wi) {
    int idx = blockIdx.x * blockDim.x + threadIdx.x;
    if (idx >= 32 * 16 * 32 * 32) return;
    int o = idx & 31, c = (idx >> 5) & 31, ky = (idx >> 10) & 15, kx = idx >> 14;
    const float* src = (kx < 16) ? w1 : w2;
    int sx = kx & 15;
    long long off = ((((long long)c * 32 + o) * 16 + sx) * 16 + ky) * 2;
    wr[idx] = (__bf16)src[off];
    wi[idx] = (__bf16)src[off + 1];
}

// fc0: concat grid, 5->32, transpose to NCHW, zero-pad to 256x256
__global__ void fc0_kernel(const float* __restrict__ x, const float* __restrict__ w,
                           const float* __restrict__ b, float* __restrict__ xpad) {
    int idx = blockIdx.x * blockDim.x + threadIdx.x;
    if (idx >= NB * NC * HW) return;
    int wi = idx & 255, h = (idx >> 8) & 255, o = (idx >> 16) & 31, bb = idx >> 21;
    float v = 0.0f;
    if (h < SX && wi < SY) {
        const float* xi = &x[(((long long)bb * SX + h) * SY + wi) * 3];
        float gx = h * (1.0f / 246.0f), gy = wi * (1.0f / 246.0f);
        const float* wr = &w[o * 5];
        v = wr[0] * xi[0] + wr[1] * xi[1] + wr[2] * xi[2] + wr[3] * gx + wr[4] * gy + b[o];
    }
    xpad[idx] = v;
}

// [b][c][kx][ky] -> [kx][ky][b][c], both complex planes
__global__ void transpose_S_kernel(const float* __restrict__ sr, const float* __restrict__ si,
                                   float* __restrict__ mr, float* __restrict__ mi) {
    int idx = blockIdx.x * blockDim.x + threadIdx.x;
    if (idx >= NB * NC * KXN * M2) return;
    int c = idx & 31, b = (idx >> 5) & 7, ky = (idx >> 8) & 15, kx = idx >> 12;
    int src = ((b * 32 + c) * 32 + kx) * 16 + ky;
    mr[idx] = sr[src]; mi[idx] = si[src];
}

// [kx][ky][b][o] -> [b][o][kx][ky]
__global__ void transpose_T_kernel(const float* __restrict__ tr, const float* __restrict__ ti,
                                   float* __restrict__ cr, float* __restrict__ ci) {
    int idx = blockIdx.x * blockDim.x + threadIdx.x;
    if (idx >= NB * NC * KXN * M2) return;
    int ky = idx & 15, kx = (idx >> 4) & 31, o = (idx >> 9) & 31, b = idx >> 14;
    int src = ((kx * 16 + ky) * 8 + b) * 32 + o;
    cr[idx] = tr[src]; ci[idx] = ti[src];
}

// CFT along h over cw -> flattened coeffs (b, (c,fh,fw,re/im))
__global__ void cft_h_kernel(const float* __restrict__ cwp, const float* __restrict__ khr,
                             const float* __restrict__ khi, float* __restrict__ flat) {
    int tid = blockIdx.x * blockDim.x + threadIdx.x;
    if (tid >= 4096) return;
    int fh = tid & 3, c = (tid >> 2) & 31, fw = (tid >> 7) & 3, b = tid >> 9;
    const float* base = &cwp[(((long long)b * 32 + c) * 256) * 8];
    float sr = 0.0f, si = 0.0f;
    for (int h = 0; h < 256; ++h) {
        float cr = base[h * 8 + fw], ci = base[h * 8 + 4 + fw];
        float kr = khr[fh * 256 + h], ki = khi[fh * 256 + h];
        sr += cr * kr - ci * ki;
        si += cr * ki + ci * kr;
    }
    int fi = ((c * 4 + fh) * 4 + fw) * 2;
    flat[b * 1024 + fi] = sr;
    flat[b * 1024 + fi + 1] = si;
}

__global__ void mlp1_kernel(const float* __restrict__ flat, const float* __restrict__ w,
                            const float* __restrict__ b, float* __restrict__ h) {
    int tid = blockIdx.x * blockDim.x + threadIdx.x;
    if (tid >= NB * 128) return;
    int bb = tid >> 7, j = tid & 127;
    float s = b[j];
    const float* f = &flat[bb * 1024];
    const float* wr = &w[(long long)j * 1024];
    for (int k = 0; k < 1024; ++k) s += f[k] * wr[k];
    h[tid] = gelu_f(s);
}

__global__ void mlp2_kernel(const float* __restrict__ h, const float* __restrict__ w,
                            const float* __restrict__ b, float* __restrict__ corr) {
    int tid = blockIdx.x * blockDim.x + threadIdx.x;
    if (tid >= NB * 32) return;
    int bb = tid >> 5, o = tid & 31;
    float s = b[o];
    const float* hp = &h[bb * 128];
    const float* wr = &w[o * 128];
    for (int j = 0; j < 128; ++j) s += hp[j] * wr[j];
    corr[tid] = s;
}

// x_next + corr broadcast (+gelu for layers 0..2) -> x_cur for next layer
__global__ void epilogue_kernel(const float* __restrict__ xn, const float* __restrict__ corr,
                                float* __restrict__ xp, int dogelu) {
    int idx = blockIdx.x * blockDim.x + threadIdx.x;
    if (idx >= NB * NC * HW) return;
    int o = (idx >> 16) & 31, b = idx >> 21;
    float v = xn[idx] + corr[b * 32 + o];
    if (dogelu) v = gelu_f(v);
    xp[idx] = v;
}

// crop + fc2 (128 -> 1)
__global__ void fc2_kernel(const __bf16* __restrict__ y1, const float* __restrict__ w,
                           const float* __restrict__ b, float* __restrict__ out) {
    int idx = blockIdx.x * blockDim.x + threadIdx.x;
    if (idx >= NB * SX * SY) return;
    int wi = idx % SY, h = (idx / SY) % SX, bb = idx / (SX * SY);
    long long base = (long long)bb * 128 * HW + h * 256 + wi;
    float s = b[0];
    for (int j = 0; j < 128; ++j)
        s += (float)y1[base + (long long)j * HW] * w[j];
    out[idx] = s;
}

// ---------------------------------------------------------------------------
// Workspace layout (float offsets, then bf16 region).  Total ~310 MB.
// ---------------------------------------------------------------------------
#define OFF_XPAD   0LL
#define OFF_XNEXT  16777216LL
#define OFF_SWR    33554432LL
#define OFF_SWI    35651584LL
#define OFF_SHR    37748736LL
#define OFF_SHI    37879808LL
#define OFF_SHMR   38010880LL
#define OFF_SHMI   38141952LL
#define OFF_TMR    38273024LL
#define OFF_TMI    38404096LL
#define OFF_TCR    38535168LL
#define OFF_TCI    38666240LL
#define OFF_UH     38797312LL
#define OFF_CWP    40894464LL
#define OFF_FLAT   41418752LL
#define OFF_HBUF   41426944LL
#define OFF_CORR   41427968LL
#define OFF_KHR    41428224LL
#define OFF_KHI    41429248LL
#define OFF_BF16   41430272LL
// bf16 element offsets within bf16 region
#define BO_FWR   0LL
#define BO_FWI   4096LL
#define BO_FHR   8192LL
#define BO_FHI   16384LL
#define BO_IFHR  24576LL
#define BO_IFHI  32768LL
#define BO_IWB   40960LL
#define BO_KW    49152LL
#define BO_WK    51200LL           // per layer: re at +L*1048576, im at +524288
#define BO_Y1    4245504LL

extern "C" void kernel_launch(void* const* d_in, const int* in_sizes, int n_in,
                              void* d_out, int out_size, void* d_ws, size_t ws_size,
                              hipStream_t stream) {
    (void)in_sizes; (void)n_in; (void)out_size; (void)ws_size;
    const float* x     = (const float*)d_in[0];
    const float* fc0_w = (const float*)d_in[1];
    const float* fc0_b = (const float*)d_in[2];
    const float* fc1_w = (const float*)d_in[3];
    const float* fc1_b = (const float*)d_in[4];
    const float* fc2_w = (const float*)d_in[5];
    const float* fc2_b = (const float*)d_in[6];

    float*  ws = (float*)d_ws;
    __bf16* bw = (__bf16*)(ws + OFF_BF16);

    float* xpad  = ws + OFF_XPAD;
    float* xnext = ws + OFF_XNEXT;
    float* Swr = ws + OFF_SWR,  * Swi = ws + OFF_SWI;
    float* Shr = ws + OFF_SHR,  * Shi = ws + OFF_SHI;
    float* Smr = ws + OFF_SHMR, * Smi = ws + OFF_SHMI;
    float* Tmr = ws + OFF_TMR,  * Tmi = ws + OFF_TMI;
    float* Tcr = ws + OFF_TCR,  * Tci = ws + OFF_TCI;
    float* Uh  = ws + OFF_UH;
    float* cwp = ws + OFF_CWP;
    float* flat = ws + OFF_FLAT;
    float* hbuf = ws + OFF_HBUF;
    float* corr = ws + OFF_CORR;
    float* Khr = ws + OFF_KHR, * Khi = ws + OFF_KHI;

    __bf16* Fwr = bw + BO_FWR,  * Fwi = bw + BO_FWI;
    __bf16* Fhr = bw + BO_FHR,  * Fhi = bw + BO_FHI;
    __bf16* IFr = bw + BO_IFHR, * IFi = bw + BO_IFHI;
    __bf16* IWB = bw + BO_IWB;
    __bf16* KW  = bw + BO_KW;
    __bf16* Y1  = bw + BO_Y1;

    // --- setup constants + pack spectral weights ---
    setup_consts_kernel<<<1, 256, 0, stream>>>(Khr, Khi, Fwr, Fwi, Fhr, Fhi,
                                               IFr, IFi, IWB, KW);
    for (int L = 0; L < 4; ++L) {
        const float* w1 = (const float*)d_in[7 + 8 * L];
        const float* w2 = (const float*)d_in[8 + 8 * L];
        pack_w_kernel<<<2048, 256, 0, stream>>>(w1, w2,
            bw + BO_WK + (long long)L * 1048576,
            bw + BO_WK + (long long)L * 1048576 + 524288);
    }

    // --- lift: fc0 + grid + pad -> xpad (NCHW) ---
    fc0_kernel<<<65536, 256, 0, stream>>>(x, fc0_w, fc0_b, xpad);

    // --- 4 spectral layers ---
    for (int L = 0; L < 4; ++L) {
        const float* g_w1 = (const float*)d_in[9  + 8 * L];
        const float* g_b1 = (const float*)d_in[10 + 8 * L];
        const float* g_w2 = (const float*)d_in[11 + 8 * L];
        const float* g_b2 = (const float*)d_in[12 + 8 * L];
        const float* wp   = (const float*)d_in[13 + 8 * L];
        const float* wb   = (const float*)d_in[14 + 8 * L];
        __bf16* WkR = bw + BO_WK + (long long)L * 1048576;
        __bf16* WkI = WkR + 524288;

        // CFT correction path (reads current activation xpad)
        gemm<float, __bf16, float>(stream, xpad, KW, cwp, nullptr,
            NB * NC * HH, 8, WW, WW, 8, 8, 0, 0, 0, 1, 1.0f, 0, 0);
        cft_h_kernel<<<16, 256, 0, stream>>>(cwp, Khr, Khi, flat);
        mlp1_kernel<<<4, 256, 0, stream>>>(flat, g_w1, g_b1, hbuf);
        mlp2_kernel<<<1, 256, 0, stream>>>(hbuf, g_w2, g_b2, corr);

        // forward DFT along w (truncated): S = x * Fw   (re / im)
        gemm<float, __bf16, float>(stream, xpad, Fwr, Swr, nullptr,
            NB * NC * HH, M2, WW, WW, M2, M2, 0, 0, 0, 1, 1.0f, 0, 0);
        gemm<float, __bf16, float>(stream, xpad, Fwi, Swi, nullptr,
            NB * NC * HH, M2, WW, WW, M2, M2, 0, 0, 0, 1, 1.0f, 0, 0);

        // forward DFT along h: Sh = Fh * S (complex), batched over (b,c)
        gemm<__bf16, float, float>(stream, Fhr, Swr, Shr, nullptr,
            KXN, M2, HH, HH, M2, M2, 0, 4096, 512, NB * NC,  1.0f, 0, 0);
        gemm<__bf16, float, float>(stream, Fhi, Swi, Shr, nullptr,
            KXN, M2, HH, HH, M2, M2, 0, 4096, 512, NB * NC, -1.0f, 1, 0);
        gemm<__bf16, float, float>(stream, Fhr, Swi, Shi, nullptr,
            KXN, M2, HH, HH, M2, M2, 0, 4096, 512, NB * NC,  1.0f, 0, 0);
        gemm<__bf16, float, float>(stream, Fhi, Swr, Shi, nullptr,
            KXN, M2, HH, HH, M2, M2, 0, 4096, 512, NB * NC,  1.0f, 1, 0);
        transpose_S_kernel<<<512, 256, 0, stream>>>(Shr, Shi, Smr, Smi);

        // per-mode channel mix (complex), batched over 512 modes, M=8 (batch dim)
        gemm<float, __bf16, float>(stream, Smr, WkR, Tmr, nullptr,
            NB, NC, NC, NC, NC, NC, 256, 1024, 256, KXN * M2,  1.0f, 0, 0);
        gemm<float, __bf16, float>(stream, Smi, WkI, Tmr, nullptr,
            NB, NC, NC, NC, NC, NC, 256, 1024, 256, KXN * M2, -1.0f, 1, 0);
        gemm<float, __bf16, float>(stream, Smr, WkI, Tmi, nullptr,
            NB, NC, NC, NC, NC, NC, 256, 1024, 256, KXN * M2,  1.0f, 0, 0);
        gemm<float, __bf16, float>(stream, Smi, WkR, Tmi, nullptr,
            NB, NC, NC, NC, NC, NC, 256, 1024, 256, KXN * M2,  1.0f, 1, 0);
        transpose_T_kernel<<<512, 256, 0, stream>>>(Tmr, Tmi, Tcr, Tci);

        // inverse along h: Uh (re at cols 0..15, im at 16..31), batched (b,o)
        gemm<__bf16, float, float>(stream, IFr, Tcr, Uh, nullptr,
            HH, M2, KXN, KXN, M2, 32, 0, 512, 8192, NB * NC,  1.0f, 0, 0);
        gemm<__bf16, float, float>(stream, IFi, Tci, Uh, nullptr,
            HH, M2, KXN, KXN, M2, 32, 0, 512, 8192, NB * NC, -1.0f, 1, 0);
        gemm<__bf16, float, float>(stream, IFr, Tci, Uh + 16, nullptr,
            HH, M2, KXN, KXN, M2, 32, 0, 512, 8192, NB * NC,  1.0f, 0, 0);
        gemm<__bf16, float, float>(stream, IFi, Tcr, Uh + 16, nullptr,
            HH, M2, KXN, KXN, M2, 32, 0, 512, 8192, NB * NC,  1.0f, 1, 0);

        // pointwise conv x2 (with bias) -> xnext, batched over b
        gemm<float, float, float>(stream, wp, xpad, xnext, wb,
            NC, HW, NC, NC, HW, HW, 0, (long long)NC * HW, (long long)NC * HW,
            NB, 1.0f, 0, 0);
        // inverse along w (rfft symmetry folded, K=32: re at k 0..15, im at k 16..31) accum
        gemm<float, __bf16, float>(stream, Uh, IWB, xnext, nullptr,
            NB * NC * HH, WW, 32, 32, WW, WW, 0, 0, 0, 1, 1.0f, 1, 0);

        // += corr broadcast, gelu (layers 0..2), becomes next xpad
        epilogue_kernel<<<65536, 256, 0, stream>>>(xnext, corr, xpad, (L < 3) ? 1 : 0);
    }

    // --- head: fc1 (WMMA, bias+gelu, bf16 out), fc2 + crop ---
    gemm<float, float, __bf16>(stream, fc1_w, xpad, Y1, fc1_b,
        128, HW, NC, NC, HW, HW, 0, (long long)NC * HW, (long long)128 * HW,
        NB, 1.0f, 0, 1);
    fc2_kernel<<<(NB * SX * SY + 255) / 256, 256, 0, stream>>>(
        Y1, fc2_w, fc2_b, (float*)d_out);
}
